// MultiHeadAttention_63806034150086
// MI455X (gfx1250) — matrix operand order repaired and checked by reading, not compiled, not run on hardware
//
#include <hip/hip_runtime.h>

// ---------------------------------------------------------------------------
// MHA forward for gfx1250 (MI455X): bf16 WMMA GEMMs + flash attention.
// TDM tensor_load_to_lds (async-to-LDS fallback) for K / W^T tile staging,
// transposed V staging, DPP-butterfly softmax max, ones-column WMMA row sums.
// B=4, S=2048, D=1024, H=16, HD=64
// ---------------------------------------------------------------------------

#define SB 4
#define SS 2048
#define SD 1024
#define SH 16
#define SHD 64

typedef __attribute__((ext_vector_type(16))) __bf16 v16bf;
typedef __attribute__((ext_vector_type(8)))  float  v8f;
typedef __attribute__((ext_vector_type(4)))  unsigned int u32x4;
typedef __attribute__((ext_vector_type(4)))  float  f32x4;
typedef __attribute__((ext_vector_type(8)))  int    i32x8;
typedef __attribute__((ext_vector_type(4)))  int    i32x4;

#if __has_builtin(__builtin_amdgcn_tensor_load_to_lds)
#define HAVE_TDM 1
#else
#define HAVE_TDM 0
#endif

union AFrag {
    v16bf v;
    unsigned short u[16];
    u32x4 q[2];
};

__device__ __forceinline__ unsigned short f2bfu(float f) {
    unsigned int u = __builtin_bit_cast(unsigned int, f);
    return (unsigned short)((u + 0x7FFFu + ((u >> 16) & 1u)) >> 16);
}

// Butterfly max over each 16-lane half (matches C-fragment row spread).
__device__ __forceinline__ float rowmax16(float x) {
#if __has_builtin(__builtin_amdgcn_mov_dpp)
    int t;
    t = __builtin_amdgcn_mov_dpp(__builtin_bit_cast(int, x), 0xB1, 0xF, 0xF, true);
    x = fmaxf(x, __builtin_bit_cast(float, t));
    t = __builtin_amdgcn_mov_dpp(__builtin_bit_cast(int, x), 0x4E, 0xF, 0xF, true);
    x = fmaxf(x, __builtin_bit_cast(float, t));
    t = __builtin_amdgcn_mov_dpp(__builtin_bit_cast(int, x), 0x141, 0xF, 0xF, true);
    x = fmaxf(x, __builtin_bit_cast(float, t));
    t = __builtin_amdgcn_mov_dpp(__builtin_bit_cast(int, x), 0x140, 0xF, 0xF, true);
    x = fmaxf(x, __builtin_bit_cast(float, t));
#else
#pragma unroll
    for (int off = 1; off < 16; off <<= 1) x = fmaxf(x, __shfl_xor(x, off, 32));
#endif
    return x;
}

#if HAVE_TDM
// Build a D# (ISA ch.8) for a 2D bf16 tile and issue TENSOR_LOAD_TO_LDS.
// tile_w elems/row (contiguous in LDS), tile_h rows, global row stride in elems.
__device__ __forceinline__ void tdm_load_2d(unsigned lds_addr, const void* gaddr,
                                            int tile_w, int tile_h,
                                            int row_stride, int tensor_rows) {
    const unsigned long long ga = (unsigned long long)gaddr;
    u32x4 g0;
    g0[0] = 1u;                                              // count=1, user D#
    g0[1] = lds_addr;                                        // LDS byte address
    g0[2] = (unsigned)(ga & 0xFFFFFFFFu);                    // global_addr[31:0]
    g0[3] = (unsigned)((ga >> 32) & 0x1FFFFFFu) | (2u << 30);// [56:32] | type=2
    i32x8 g1;
    g1[0] = 1 << 16;                                         // data_size=1 (2B)
    g1[1] = (row_stride & 0xFFFF) << 16;                     // tensor_dim0 lo
    g1[2] = ((tensor_rows & 0xFFFF) << 16) | ((row_stride >> 16) & 0xFFFF);
    g1[3] = ((tile_w & 0xFFFF) << 16) | ((tensor_rows >> 16) & 0xFFFF);
    g1[4] = tile_h & 0xFFFF;                                 // tile_dim1 (tile_dim2=0)
    g1[5] = row_stride;                                      // tensor_dim0_stride lo
    g1[6] = 0;
    g1[7] = 0;
    const i32x4 z4 = {0, 0, 0, 0};
    const i32x8 z8 = {0, 0, 0, 0, 0, 0, 0, 0};
    __builtin_amdgcn_tensor_load_to_lds(g0, g1, z4, z4, z8, 0);
}
#endif

// Stage a tile_h x tile_w bf16 tile (global row stride row_stride elems) into
// contiguous LDS. TDM: single instruction from one wave. Fallback: per-lane
// async-to-LDS b128 copies spread over nthr threads.
__device__ __forceinline__ void stage_tile(const unsigned short* g, unsigned short* lds,
                                           int tile_w, int tile_h, int row_stride,
                                           int tensor_rows, int t, int nthr, bool issuer) {
#if HAVE_TDM
    if (issuer)
        tdm_load_2d((unsigned)(unsigned long long)lds, g, tile_w, tile_h,
                    row_stride, tensor_rows);
#else
    const int cpr = tile_w >> 3;            // 16B chunks per row
    const int nchunks = tile_h * cpr;
    for (int c = t; c < nchunks; c += nthr) {
        const int row = c / cpr;
        const int off = (c % cpr) * 8;
        const unsigned short* gp = g + (size_t)row * row_stride + off;
        const unsigned la = (unsigned)(unsigned long long)(lds + row * tile_w + off);
        asm volatile("global_load_async_to_lds_b128 %0, %1, off"
                     :: "v"(la), "v"((unsigned long long)gp) : "memory");
    }
#endif
}

__device__ __forceinline__ void stage_wait(bool issuer) {
#if HAVE_TDM
    if (issuer) {
#if __has_builtin(__builtin_amdgcn_s_wait_tensorcnt)
        __builtin_amdgcn_s_wait_tensorcnt(0);
#else
        asm volatile("s_wait_tensorcnt 0" ::: "memory");
#endif
    }
#else
#if __has_builtin(__builtin_amdgcn_s_wait_asynccnt)
    __builtin_amdgcn_s_wait_asynccnt(0);
#else
    asm volatile("s_wait_asynccnt 0" ::: "memory");
#endif
#endif
}

// ---------------------------------------------------------------------------
// Transpose+convert one 1024x1024 f32 weight into bf16 W^T[N][K].
// ---------------------------------------------------------------------------
__global__ void __launch_bounds__(256)
transpose_to_bf16(const float* __restrict__ in, unsigned short* __restrict__ out) {
    const int tid = blockIdx.x * 256 + threadIdx.x;
    const int n  = tid & 1023;
    const int k0 = (tid >> 10) * 8;
    unsigned short tmp[8];
#pragma unroll
    for (int i = 0; i < 8; ++i) tmp[i] = f2bfu(in[(size_t)(k0 + i) * SD + n]);
    u32x4 pk;
#pragma unroll
    for (int i = 0; i < 4; ++i)
        pk[i] = (unsigned)tmp[2 * i] | ((unsigned)tmp[2 * i + 1] << 16);
    *(u32x4*)(out + (size_t)n * SD + k0) = pk;
}

// ---------------------------------------------------------------------------
// Tiled GEMM: C[M x 1024] = A[M x 1024] @ W, with W given as bf16 W^T[N][K].
// Block: 256 threads = 8 waves; wave strip 16(M) x 64(N).
// W^T slab (64 n-rows x 32 k) staged to LDS via TDM/async, double buffered.
// ---------------------------------------------------------------------------
template <bool IN_BF16, bool OUT_F32>
__global__ void __launch_bounds__(256)
gemm_bf16_wmma(const void* __restrict__ xv, const unsigned short* __restrict__ wt,
               void* __restrict__ outv) {
    __shared__ unsigned short wlds[2][64 * 32];  // 2 x 4 KB W^T tiles

    const int t    = threadIdx.x;
    const int lane = t & 31;
    const int wave = t >> 5;
    const int mbase = blockIdx.x * 128 + wave * 16;
    const int n0 = blockIdx.y * 64;
    const bool issuer = (wave == 0);

    const int mrow = lane & 15;
    const int hi   = lane >> 4;
    const int akb  = hi ? 8 : 0;
    const int bkb  = hi ? 8 : 0;

    const float*          xf = (const float*)xv;
    const unsigned short* xb = (const unsigned short*)xv;

    v8f acc[4] = {};

    // prologue: stage k0 = 0 slab
    stage_tile(wt + (size_t)n0 * SD, wlds[0], 32, 64, SD, SD, t, 256, issuer);
    stage_wait(issuer);
    __syncthreads();

    int cur = 0;
    for (int k0 = 0; k0 < SD; k0 += 32) {
        const bool hasnext = (k0 + 32) < SD;
        if (hasnext)
            stage_tile(wt + (size_t)n0 * SD + (k0 + 32), wlds[cur ^ 1],
                       32, 64, SD, SD, t, 256, issuer);

        AFrag a;
        if (IN_BF16) {
            const unsigned short* r = xb + (size_t)(mbase + mrow) * SD + k0 + akb;
            a.q[0] = *(const u32x4*)(r);
            a.q[1] = *(const u32x4*)(r + 16);
        } else {
            const float* r = xf + (size_t)(mbase + mrow) * SD + k0 + akb;
            f32x4 f0 = *(const f32x4*)(r);
            f32x4 f1 = *(const f32x4*)(r + 4);
            f32x4 f2 = *(const f32x4*)(r + 16);
            f32x4 f3 = *(const f32x4*)(r + 20);
#pragma unroll
            for (int i = 0; i < 4; ++i) {
                a.u[i]      = f2bfu(f0[i]);
                a.u[4 + i]  = f2bfu(f1[i]);
                a.u[8 + i]  = f2bfu(f2[i]);
                a.u[12 + i] = f2bfu(f3[i]);
            }
        }

#pragma unroll
        for (int nt = 0; nt < 4; ++nt) {
            AFrag b;  // (k,n) = W[k0+k, n0+nt*16+n] = wtlds[(nt*16+n)*32 + k]
            const unsigned short* r = &wlds[cur][(nt * 16 + mrow) * 32 + bkb];
            b.q[0] = *(const u32x4*)(r);
            b.q[1] = *(const u32x4*)(r + 16);
            acc[nt] = __builtin_amdgcn_wmma_f32_16x16x32_bf16(
                false, a.v, false, b.v, (short)0, acc[nt], false, false);
        }

        if (hasnext) stage_wait(issuer);
        __syncthreads();
        cur ^= 1;
    }

#pragma unroll
    for (int nt = 0; nt < 4; ++nt) {
#pragma unroll
        for (int v = 0; v < 8; ++v) {
            const int row = mbase + v + 8 * hi;
            const int col = n0 + nt * 16 + mrow;
            const float val = acc[nt][v];
            if (OUT_F32)
                ((float*)outv)[(size_t)row * SD + col] = val;
            else
                ((unsigned short*)outv)[(size_t)row * SD + col] = f2bfu(val);
        }
    }
}

// ---------------------------------------------------------------------------
// Flash attention. Block = 128 threads = 4 waves sharing one (b,h); each wave
// owns a 16-row Q tile. K tiles staged via TDM/async (double buffered, one
// tile ahead); V tiles staged transposed via registers (double buffered).
// ---------------------------------------------------------------------------
__device__ __forceinline__ void loadV_regs(const unsigned short* gbase, int kt, int t,
                                           AFrag& r) {
    const int key = t >> 2;
    const int hd0 = (t & 3) * 16;
    const unsigned short* g = gbase + (size_t)(kt + key) * SD + hd0;
    r.q[0] = *(const u32x4*)(g);
    r.q[1] = *(const u32x4*)(g + 8);
}

__device__ __forceinline__ void storeV_T(unsigned short* vt, int t, const AFrag& r) {
    const int key = t >> 2;
    const int hd0 = (t & 3) * 16;
#pragma unroll
    for (int i = 0; i < 16; ++i) vt[(hd0 + i) * 32 + key] = r.u[i];
}

__global__ void __launch_bounds__(128)
flash_attn_wmma(const unsigned short* __restrict__ qb,
                const unsigned short* __restrict__ kb,
                const unsigned short* __restrict__ vb,
                const float* __restrict__ mask,
                unsigned short* __restrict__ ob) {
    __shared__ unsigned short klds[2][32 * 64];  // 2 x 4 KB K tiles
    __shared__ unsigned short vtld[2][64 * 32];  // 2 x 4 KB V^T tiles
    __shared__ unsigned short plds[4][16 * 32];  // per-wave P tiles

    const int t    = threadIdx.x;
    const int lane = t & 31;
    const int wave = t >> 5;
    const int bh = blockIdx.y;
    const int b = bh >> 4;
    const int h = bh & 15;
    const int qbase = (blockIdx.x * 4 + wave) * 16;
    const bool issuer = (wave == 0);

    const int mrow = lane & 15;
    const int hi   = lane >> 4;
    const int akb  = hi ? 8 : 0;
    const int bkb  = hi ? 8 : 0;

    const unsigned short* qptr = qb + (size_t)b * SS * SD + h * SHD;
    const unsigned short* kptr = kb + (size_t)b * SS * SD + h * SHD;
    const unsigned short* vptr = vb + (size_t)b * SS * SD + h * SHD;

    AFrag aq[2];
#pragma unroll
    for (int c = 0; c < 2; ++c) {
        const unsigned short* r = qptr + (size_t)(qbase + mrow) * SD + c * 32 + akb;
        aq[c].q[0] = *(const u32x4*)(r);
        aq[c].q[1] = *(const u32x4*)(r + 16);
    }

    AFrag bone;
#pragma unroll
    for (int j = 0; j < 16; ++j) bone.u[j] = 0x3F80;  // bf16 1.0

    float m[8];
    v8f o[4] = {};
    v8f osum = {};
#pragma unroll
    for (int v = 0; v < 8; ++v) m[v] = -1e30f;

    const float scale = 0.125f;  // 1/sqrt(64)

    // ---- prologue: stage tile 0
    stage_tile(kptr, klds[0], 64, 32, SD, SS, t, 128, issuer);
    {
        AFrag vr;
        loadV_regs(vptr, 0, t, vr);
        storeV_T(vtld[0], t, vr);
    }
    stage_wait(issuer);
    __syncthreads();

    int cur = 0;
    for (int kt = 0; kt < SS; kt += 32) {
        const bool hasnext = (kt + 32) < SS;
        AFrag vr;
        if (hasnext) {
            loadV_regs(vptr, kt + 32, t, vr);  // fire early, consumed late
            stage_tile(kptr + (size_t)(kt + 32) * SD, klds[cur ^ 1],
                       64, 32, SD, SS, t, 128, issuer);
        }

        // ---- S = Q @ K^T from LDS K tile
        v8f s[2];
#pragma unroll
        for (int nt = 0; nt < 2; ++nt) {
            v8f acc = {};
#pragma unroll
            for (int c = 0; c < 2; ++c) {
                AFrag bk;
                const unsigned short* r = &klds[cur][(nt * 16 + mrow) * 64 + c * 32 + bkb];
                bk.q[0] = *(const u32x4*)(r);
                bk.q[1] = *(const u32x4*)(r + 16);
                acc = __builtin_amdgcn_wmma_f32_16x16x32_bf16(
                    false, aq[c].v, false, bk.v, (short)0, acc, false, false);
            }
            s[nt] = acc;
        }

        // ---- online softmax (max via DPP; sum via ones-column WMMA)
#pragma unroll
        for (int v = 0; v < 8; ++v) {
            const int row = qbase + v + 8 * hi;
            const float s0 = s[0][v] * scale + mask[(size_t)row * SS + kt + mrow];
            const float s1 = s[1][v] * scale + mask[(size_t)row * SS + kt + 16 + mrow];
            const float mn = fmaxf(m[v], rowmax16(fmaxf(s0, s1)));
            const float alpha = __expf(m[v] - mn);
            m[v] = mn;
            const float p0 = __expf(s0 - mn);
            const float p1 = __expf(s1 - mn);
            osum[v] = osum[v] * alpha;
#pragma unroll
            for (int n = 0; n < 4; ++n)
                o[n][v] = o[n][v] * alpha;
            plds[wave][(v + 8 * hi) * 32 + mrow]      = f2bfu(p0);
            plds[wave][(v + 8 * hi) * 32 + 16 + mrow] = f2bfu(p1);
        }
        __syncthreads();

        AFrag ap;
        {
            const unsigned short* r = &plds[wave][mrow * 32 + akb];
            ap.q[0] = *(const u32x4*)(r);
            ap.q[1] = *(const u32x4*)(r + 16);
        }

        // ---- O += P @ V ; row sums via ones column
#pragma unroll
        for (int nt = 0; nt < 4; ++nt) {
            AFrag bv;
            const unsigned short* r = &vtld[cur][(nt * 16 + mrow) * 32 + bkb];
            bv.q[0] = *(const u32x4*)(r);
            bv.q[1] = *(const u32x4*)(r + 16);
            o[nt] = __builtin_amdgcn_wmma_f32_16x16x32_bf16(
                false, ap.v, false, bv.v, (short)0, o[nt], false, false);
        }
        osum = __builtin_amdgcn_wmma_f32_16x16x32_bf16(
            false, ap.v, false, bone.v, (short)0, osum, false, false);

        if (hasnext) {
            storeV_T(vtld[cur ^ 1], t, vr);
            stage_wait(issuer);
        }
        __syncthreads();
        cur ^= 1;
    }

#pragma unroll
    for (int nt = 0; nt < 4; ++nt) {
#pragma unroll
        for (int v = 0; v < 8; ++v) {
            const int row = qbase + v + 8 * hi;
            const float val = o[nt][v] / osum[v];
            ob[((size_t)b * SS + row) * SD + h * SHD + nt * 16 + mrow] = f2bfu(val);
        }
    }
}

// ---------------------------------------------------------------------------
// Host-side launch
// inputs: 0=q 1=k 2=v 3=mask 4=wq 5=wk 6=wv 7=wo   (all fp32)
// ---------------------------------------------------------------------------
extern "C" void kernel_launch(void* const* d_in, const int* in_sizes, int n_in,
                              void* d_out, int out_size, void* d_ws, size_t ws_size,
                              hipStream_t stream) {
    const float* q    = (const float*)d_in[0];
    const float* k    = (const float*)d_in[1];
    const float* v    = (const float*)d_in[2];
    const float* mask = (const float*)d_in[3];
    const float* wq   = (const float*)d_in[4];
    const float* wk   = (const float*)d_in[5];
    const float* wv   = (const float*)d_in[6];
    const float* wo   = (const float*)d_in[7];

    const size_t elems  = (size_t)SB * SS * SD;   // 8,388,608
    const size_t welems = (size_t)SD * SD;        // 1,048,576
    unsigned short* q_buf = (unsigned short*)d_ws;
    unsigned short* k_buf = q_buf + elems;
    unsigned short* v_buf = k_buf + elems;
    unsigned short* a_buf = v_buf + elems;
    unsigned short* wt_q  = a_buf + elems;
    unsigned short* wt_k  = wt_q + welems;
    unsigned short* wt_v  = wt_k + welems;
    unsigned short* wt_o  = wt_v + welems;

    // weights -> bf16 W^T
    transpose_to_bf16<<<512, 256, 0, stream>>>(wq, wt_q);
    transpose_to_bf16<<<512, 256, 0, stream>>>(wk, wt_k);
    transpose_to_bf16<<<512, 256, 0, stream>>>(wv, wt_v);
    transpose_to_bf16<<<512, 256, 0, stream>>>(wo, wt_o);

    const dim3 gblk(256);
    const dim3 ggrd((SB * SS) / 128, SD / 64);

    gemm_bf16_wmma<false, false><<<ggrd, gblk, 0, stream>>>(q, wt_q, q_buf);
    gemm_bf16_wmma<false, false><<<ggrd, gblk, 0, stream>>>(k, wt_k, k_buf);
    gemm_bf16_wmma<false, false><<<ggrd, gblk, 0, stream>>>(v, wt_v, v_buf);

    flash_attn_wmma<<<dim3(SS / 64, SB * SH), dim3(128), 0, stream>>>(
        q_buf, k_buf, v_buf, mask, a_buf);

    gemm_bf16_wmma<true, true><<<ggrd, gblk, 0, stream>>>(a_buf, wt_o, (float*)d_out);
}
